// MyRNN_58480274703119
// MI455X (gfx1250) — matrix operand order repaired and checked by reading, not compiled, not run on hardware
//
#include <hip/hip_runtime.h>

// MI455X / gfx1250: wave32, WMMA 16x16x32 bf16 (f32 accumulate),
// persistent kernel + LDS-resident weights + device-wide per-step barrier.

typedef __attribute__((ext_vector_type(16))) __bf16 v16bf;
typedef __attribute__((ext_vector_type(8)))  float  v8f;
typedef __attribute__((ext_vector_type(4)))  float  v4f;

#define B_    64
#define T_    512
#define D_    128
#define H_    1024
#define NBLK  32      // persistent grid: 32 workgroups x 256 threads = 256 waves

// ---------------------------------------------------------------------------
// Pack fp32 weight matrix W[K x H] (row-major) into bf16 in WMMA B-fragment
// order: linear index = ((nt*KB + kb)*32 + lane)*16 + i  holds
//   W[kb*32 + (lane>>4)*16 + i][nt*16 + (lane&15)]
// so at GEMM time each lane does ONE contiguous 32-byte load per K=32 block.
// ---------------------------------------------------------------------------
__global__ void pack_w_bf16(const float* __restrict__ W, __bf16* __restrict__ dst,
                            int K, int Hn) {
  int idx = blockIdx.x * blockDim.x + threadIdx.x;
  int total = K * Hn;
  if (idx >= total) return;
  int KB   = K >> 5;
  int i    = idx & 15;
  int lane = (idx >> 4) & 31;
  int kb   = (idx >> 9) % KB;
  int nt   = (idx >> 9) / KB;
  int srcK = kb * 32 + ((lane >> 4) << 4) + i;
  int srcN = nt * 16 + (lane & 15);
  dst[idx] = (__bf16)W[srcK * Hn + srcN];
}

__global__ void zero_sync(int* __restrict__ sync) {
  int i = blockIdx.x * blockDim.x + threadIdx.x;
  if (i < T_) sync[i] = 0;
}

// ---------------------------------------------------------------------------
// 16x32 bf16 A-fragment from an fp32 row-major source row.
// Layout used here (k contiguous, as packed B): lanes 0-15 -> M=lane, K=0..15
//                          lanes 16-31 -> M=lane-16, K=16..31
// ---------------------------------------------------------------------------
__device__ __forceinline__ v16bf load_a_frag(const float* __restrict__ rowPtr,
                                             int k0, int kOffA) {
  const v4f* p = (const v4f*)(rowPtr + k0 + kOffA);
  v4f f0 = p[0];
  v4f f1 = p[1];
  const v4f* q = (const v4f*)(rowPtr + k0 + kOffA + 8);
  v4f f2 = q[0];
  v4f f3 = q[1];
  v16bf a;
#pragma unroll
  for (int j = 0; j < 4; ++j) {
    a[j]      = (__bf16)f0[j];
    a[4 + j]  = (__bf16)f1[j];
    a[8 + j]  = (__bf16)f2[j];
    a[12 + j] = (__bf16)f3[j];
  }
  return a;
}

// ---------------------------------------------------------------------------
// Persistent RNN kernel. Block b covers N-tiles {2b, 2b+1} x all 4 M-tiles:
//   wave w (0..7): mt = w & 3, nt = 2*blockIdx.x + (w >> 2)
// Wh/Wx B-fragments for the block's two N-tiles are staged in LDS once.
// Per step:  out[:,t,:] = tanh( x[:,t,:]@Wx + b + out[:,t-1,:]@Wh )
// followed by a device-wide barrier (atomic arrive + spin, s_sleep backoff).
// ---------------------------------------------------------------------------
__global__ __launch_bounds__(256)
void rnn_persist(const float* __restrict__ x, const __bf16* __restrict__ WxP,
                 const __bf16* __restrict__ WhP, const float* __restrict__ bias,
                 float* __restrict__ out, int* __restrict__ sync) {
  __shared__ __bf16 lWh[2 * 32 * 32 * 16];   // 64 KB: [ti][kb][lane][16]
  __shared__ __bf16 lWx[2 * 4 * 32 * 16];    //  8 KB: [ti][kb][lane][16]

  const int lane   = threadIdx.x & 31;
  const int w      = threadIdx.x >> 5;
  const int mt     = w & 3;                  // 0..3
  const int ti     = w >> 2;                 // 0..1
  const int nt     = blockIdx.x * 2 + ti;    // 0..63
  const int laneHi = lane >> 4;
  const int rowA   = lane & 15;
  const int kOffA  = laneHi * 16;
  const int nCol   = nt * 16 + (lane & 15);
  const int batchA = mt * 16 + rowA;

  // ---- stage weight B-fragments into LDS (once) ----
#pragma unroll
  for (int s = w * 8; s < w * 8 + 8; ++s) {  // 64 Wh (ti,kb) slices / 8 waves
    int sti = s >> 5, skb = s & 31;
    int gnt = blockIdx.x * 2 + sti;
    v16bf v = *(const v16bf*)(WhP + (((size_t)gnt * 32 + skb) * 32 + lane) * 16);
    *(v16bf*)&lWh[(((size_t)sti * 32 + skb) * 32 + lane) * 16] = v;
  }
  {                                          // 8 Wx slices / 8 waves
    int sti = w >> 2, skb = w & 3;
    int gnt = blockIdx.x * 2 + sti;
    v16bf v = *(const v16bf*)(WxP + (((size_t)gnt * 4 + skb) * 32 + lane) * 16);
    *(v16bf*)&lWx[(((size_t)sti * 4 + skb) * 32 + lane) * 16] = v;
  }
  __syncthreads();

  const float bn = bias[nCol];

  for (int t = 0; t < T_; ++t) {
    v8f acc = {};

    // ---- x_t @ Wx  (K = 128, 4 WMMAs, B from LDS) ----
    {
      const float* xRow = x + ((size_t)batchA * T_ + t) * D_;
#pragma unroll
      for (int kb = 0; kb < 4; ++kb) {
        v16bf a  = load_a_frag(xRow, kb * 32, kOffA);
        v16bf bf = *(const v16bf*)&lWx[(((size_t)ti * 4 + kb) * 32 + lane) * 16];
        acc = __builtin_amdgcn_wmma_f32_16x16x32_bf16(
            false, a, false, bf, (short)0, acc, false, false);
      }
    }

    // ---- h_{t-1} @ Wh  (K = 1024, 32 WMMAs, B from LDS); h_{-1} == 0 ----
    if (t > 0) {
      const float* hRow = out + ((size_t)batchA * T_ + (t - 1)) * H_;
#pragma unroll 4
      for (int kb = 0; kb < 32; ++kb) {
        v16bf a  = load_a_frag(hRow, kb * 32, kOffA);
        v16bf bf = *(const v16bf*)&lWh[(((size_t)ti * 32 + kb) * 32 + lane) * 16];
        acc = __builtin_amdgcn_wmma_f32_16x16x32_bf16(
            false, a, false, bf, (short)0, acc, false, false);
      }
    }

    // ---- epilogue: + bias, tanh, store h_t into out[:,t,:] ----
    // C/D layout: lane l, VGPR r -> (M = r + 8*(l>>4), N = l&15)
#pragma unroll
    for (int r = 0; r < 8; ++r) {
      int bM = mt * 16 + r + 8 * laneHi;
      out[((size_t)bM * T_ + t) * H_ + nCol] = tanhf(acc[r] + bn);
    }

    // ---- device-wide barrier: h_t visible to all blocks before step t+1 ----
    __threadfence();                          // release h_t stores (agent scope)
    __syncthreads();
    if (threadIdx.x == 0) {
      __hip_atomic_fetch_add(&sync[t], 1, __ATOMIC_RELEASE,
                             __HIP_MEMORY_SCOPE_AGENT);
      while (__hip_atomic_load(&sync[t], __ATOMIC_ACQUIRE,
                               __HIP_MEMORY_SCOPE_AGENT) < NBLK) {
        __builtin_amdgcn_s_sleep(8);
      }
    }
    __syncthreads();
    __threadfence();                          // acquire other blocks' h_t
  }
}

// ---------------------------------------------------------------------------
extern "C" void kernel_launch(void* const* d_in, const int* in_sizes, int n_in,
                              void* d_out, int out_size, void* d_ws, size_t ws_size,
                              hipStream_t stream) {
  const float* x    = (const float*)d_in[0];  // [64,512,128]
  const float* Wx   = (const float*)d_in[1];  // [128,1024]
  const float* Wh   = (const float*)d_in[2];  // [1024,1024]
  const float* bias = (const float*)d_in[3];  // [1024]
  float* out = (float*)d_out;                 // [64,512,1024]

  __bf16* WxP = (__bf16*)d_ws;                          // 256 KB
  __bf16* WhP = WxP + (size_t)D_ * H_;                  // 2 MB
  int*    syn = (int*)((char*)d_ws +
                       ((size_t)D_ * H_ + (size_t)H_ * H_) * sizeof(__bf16));

  // Re-init per launch (deterministic across graph replays).
  zero_sync<<<2, 256, 0, stream>>>(syn);
  pack_w_bf16<<<(D_ * H_ + 255) / 256, 256, 0, stream>>>(Wx, WxP, D_, H_);
  pack_w_bf16<<<(H_ * H_ + 255) / 256, 256, 0, stream>>>(Wh, WhP, H_, H_);

  // One persistent kernel runs the whole 512-step recurrence.
  rnn_persist<<<dim3(NBLK, 1, 1), 256, 0, stream>>>(x, WxP, WhP, bias, out, syn);
}
